// Decoder_45380624450003
// MI455X (gfx1250) — compile-verified, matrix operand order checked by reading, not run on hardware
//
#include <hip/hip_runtime.h>
#include <hip/hip_bf16.h>

typedef _Float16 half_t;
typedef __attribute__((ext_vector_type(16))) _Float16 v16h;
typedef __attribute__((ext_vector_type(8)))  _Float16 v8h;
typedef __attribute__((ext_vector_type(8)))  float    v8f;
typedef __attribute__((ext_vector_type(4)))  unsigned int u32x4;
typedef __attribute__((ext_vector_type(8)))  int          i32x8;
typedef __attribute__((ext_vector_type(4)))  int          i32x4;

#define WMMA_F16(A, B, C) \
    __builtin_amdgcn_wmma_f32_16x16x32_f16(false, (A), false, (B), (short)0, (C), false, false)

__device__ __forceinline__ v16h make_a(v8h lo, v8h hi) {
    v16h r;
#pragma unroll
    for (int i = 0; i < 8; ++i) { r[i] = lo[i]; r[i + 8] = hi[i]; }
    return r;
}

// ---------------------------------------------------------------------------
// TDM: async 2D f16 tile load Global -> LDS (Tensor Data Mover, TENSORcnt).
// Descriptor bit layout per CDNA5 ISA ch.8 (group0/group1; groups 2-3 zero
// for 2D tensors).  tile = tile_d0 x tile_d1 elements, row stride = stride0
// elements, contiguous rows of tile_d0*2 bytes in LDS starting at lds_off.
// ---------------------------------------------------------------------------
__device__ __forceinline__ void tdm_load_2d_f16(
    unsigned lds_off, const half_t* gptr, unsigned tile_d0, unsigned tile_d1,
    unsigned tensor_d0, unsigned tensor_d1, unsigned stride0) {
    unsigned long long ga = (unsigned long long)(size_t)gptr;
    u32x4 g0;
    g0[0] = 1u;                                       // count=1 (valid), load, user
    g0[1] = lds_off;                                  // lds_addr[31:0]
    g0[2] = (unsigned)(ga & 0xFFFFFFFFu);             // global_addr[31:0]
    g0[3] = (unsigned)((ga >> 32) & 0x01FFFFFFu)      // global_addr[56:32]
          | (2u << 30);                               // type=2 ("image")
    i32x8 g1;
    g1[0] = (int)(1u << 16);                          // data_size=1 (2 bytes)
    g1[1] = (int)((tensor_d0 & 0xFFFFu) << 16);       // tensor_dim0[15:0] @ bit48
    g1[2] = (int)((tensor_d0 >> 16) | ((tensor_d1 & 0xFFFFu) << 16));
    g1[3] = (int)((tensor_d1 >> 16) | (tile_d0 << 16));  // tile_dim0 @ bit112
    g1[4] = (int)(tile_d1 & 0xFFFFu);                 // tile_dim1 (tile_dim2=0)
    g1[5] = (int)stride0;                             // tensor_dim0_stride[31:0]
    g1[6] = 0;
    g1[7] = 0;                                        // tensor_dim1_stride=0 (2D)
    i32x4 z4 = {0, 0, 0, 0};
#if __clang_major__ >= 23
    i32x8 z8v = {0, 0, 0, 0, 0, 0, 0, 0};
    __builtin_amdgcn_tensor_load_to_lds(g0, g1, z4, z4, z8v, 0);
#else
    __builtin_amdgcn_tensor_load_to_lds(g0, g1, z4, z4, 0);
#endif
}

// ---------------------------------------------------------------------------
// Kernel 0: pack 5 weight matrices (128x128 f32, row-major W[k][n]) into the
// WMMA B-operand lane layout as f16 so a wave loads one v16h (32B) per tile.
// ---------------------------------------------------------------------------
__global__ __launch_bounds__(256) void pack_weights(
    const float* __restrict__ Wq1, const float* __restrict__ Wq0,
    const float* __restrict__ Wk,  const float* __restrict__ Wv,
    const float* __restrict__ Wc,  half_t* __restrict__ dst) {
    int idx = blockIdx.x * 256 + threadIdx.x;          // < 5*16384 = 81920
    int w   = idx >> 14, rem = idx & 16383;
    int t   = rem >> 9,  le  = rem & 511;
    int lane = le >> 4,  e   = le & 15;
    int ntile = t >> 2,  ktile = t & 3;
    int k = ktile * 32 + ((lane >> 4) << 3) + (e < 8 ? e : e + 8);
    int n = ntile * 16 + (lane & 15);
    const float* W = (w == 0) ? Wq1 : (w == 1) ? Wq0 : (w == 2) ? Wk : (w == 3) ? Wv : Wc;
    dst[idx] = (half_t)W[k * 128 + n];
}

// ---------------------------------------------------------------------------
// Kernel 1: projections.  One block per 16 flat rows of (B*N, 128).
// K = nodes@Wk, V = nodes@Wv, Q = 0.25*(q1@Wq1 + q0@Wq0); also nodes -> f16.
// ---------------------------------------------------------------------------
__global__ __launch_bounds__(256) void proj_kernel(
    const float* __restrict__ nodes, const float* __restrict__ q1,
    const float* __restrict__ q0,
    const half_t* __restrict__ pWk,  const half_t* __restrict__ pWv,
    const half_t* __restrict__ pWq1, const half_t* __restrict__ pWq0,
    half_t* __restrict__ Kh, half_t* __restrict__ Vh,
    half_t* __restrict__ Qh, half_t* __restrict__ Nh) {
    __shared__ __align__(16) half_t sN[16][136], sQ1[16][136], sQ0[16][136];
    const int r0 = blockIdx.x * 16;
    for (int i = threadIdx.x; i < 2048; i += 256) {
        int rr = i >> 7, cc = i & 127;
        size_t g = (size_t)(r0 + rr) * 128 + cc;
        half_t a = (half_t)nodes[g];
        sN[rr][cc]  = a;  Nh[g] = a;
        sQ1[rr][cc] = (half_t)q1[g];
        sQ0[rr][cc] = (half_t)q0[g];
    }
    __syncthreads();
    const int lane = threadIdx.x & 31, w = threadIdx.x >> 5;
    const int row = lane & 15, kb = (lane >> 4) << 3;
    v8f accK = {}, accV = {}, accQ = {};
#pragma unroll
    for (int kc = 0; kc < 4; ++kc) {
        const int k0 = kc * 32;
        v16h aN  = make_a(*(const v8h*)&sN[row][k0 + kb],  *(const v8h*)&sN[row][k0 + kb + 16]);
        v16h aQ1 = make_a(*(const v8h*)&sQ1[row][k0 + kb], *(const v8h*)&sQ1[row][k0 + kb + 16]);
        v16h aQ0 = make_a(*(const v8h*)&sQ0[row][k0 + kb], *(const v8h*)&sQ0[row][k0 + kb + 16]);
        const int t = (w * 4 + kc) * 512 + lane * 16;
        v16h bK = *(const v16h*)(pWk  + t);
        v16h bV = *(const v16h*)(pWv  + t);
        v16h b1 = *(const v16h*)(pWq1 + t);
        v16h b0 = *(const v16h*)(pWq0 + t);
        accK = WMMA_F16(aN,  bK, accK);
        accV = WMMA_F16(aN,  bV, accV);
        accQ = WMMA_F16(aQ1, b1, accQ);
        accQ = WMMA_F16(aQ0, b0, accQ);
    }
    const int col = w * 16 + (lane & 15);
#pragma unroll
    for (int r = 0; r < 8; ++r) {
        size_t g = (size_t)(r0 + r + ((lane >> 4) << 3)) * 128 + col;
        Kh[g] = (half_t)accK[r];
        Vh[g] = (half_t)accV[r];
        Qh[g] = (half_t)(accQ[r] * 0.25f);   // fold 1/sqrt(D) into Q
    }
}

// ---------------------------------------------------------------------------
// Kernel 2: fused multi-head attention (flash-style online softmax).
// One block per (b,h); K and Q tiles staged by the Tensor Data Mover
// (async, TENSORcnt), V transposed manually; 8 waves over query tiles.
// ---------------------------------------------------------------------------
__global__ __launch_bounds__(256) void attn_kernel(
    const half_t* __restrict__ Kh, const half_t* __restrict__ Vh,
    const half_t* __restrict__ Qh, const float* __restrict__ mask,
    half_t* __restrict__ outc) {
    __shared__ __align__(16) half_t sK[512][16];     // [key][d]
    __shared__ __align__(16) half_t sVT[16][520];    // [d][key]
    __shared__ __align__(16) half_t sQ[512][16];     // [query][d]
    __shared__ __align__(16) half_t sP[8][16][40];   // per-wave P repack scratch
    const int b = blockIdx.x >> 3, h = blockIdx.x & 7;

    // TDM: rows 0..499 of sK/sQ come from 500x16 f16 tiles (row stride 128).
    if (threadIdx.x < 32) {
        const size_t base = (size_t)(b * 500) * 128 + h * 16;
        tdm_load_2d_f16((unsigned)(size_t)&sK[0][0], Kh + base, 16, 500, 128, 500, 128);
        tdm_load_2d_f16((unsigned)(size_t)&sQ[0][0], Qh + base, 16, 500, 128, 500, 128);
    }
    // Manual staging: V transposed; zero the pad rows 500..511 (disjoint from TDM).
    for (int m = threadIdx.x; m < 512; m += 256) {
        if (m < 500) {
            const size_t base = (size_t)(b * 500 + m) * 128 + h * 16;
#pragma unroll
            for (int d = 0; d < 16; ++d) sVT[d][m] = Vh[base + d];
        } else {
#pragma unroll
            for (int d = 0; d < 16; ++d) {
                sK[m][d] = (half_t)0; sQ[m][d] = (half_t)0; sVT[d][m] = (half_t)0;
            }
        }
    }
    if (threadIdx.x < 32) __builtin_amdgcn_s_wait_tensorcnt(0);
    __syncthreads();

    const int lane = threadIdx.x & 31, w = threadIdx.x >> 5;
    const int cr = lane & 15, hi = lane >> 4, kb = hi << 3;
    v8h z8 = {};
    for (int mt = w; mt < 32; mt += 8) {
        const int n0 = mt * 16;
        v16h aQ = make_a(*(const v8h*)&sQ[n0 + cr][kb], z8);   // K(d) padded to 32
        v8f acc = {};
        float mrun[8], lrun[8];
#pragma unroll
        for (int r = 0; r < 8; ++r) { mrun[r] = -1e30f; lrun[r] = 0.f; }
        for (int c = 0; c < 16; ++c) {
            const int kc = c * 32;
            v16h bK1 = make_a(*(const v8h*)&sK[kc + cr][kb],      z8);
            v16h bK2 = make_a(*(const v8h*)&sK[kc + 16 + cr][kb], z8);
            v8f s1 = {}, s2 = {};
            s1 = WMMA_F16(aQ, bK1, s1);
            s2 = WMMA_F16(aQ, bK2, s2);
            float v1[8], v2[8];
            const int k1 = kc + cr, k2 = kc + 16 + cr;
#pragma unroll
            for (int r = 0; r < 8; ++r) {
                const int rg = n0 + r + (hi << 3);
                float m1 = (rg < 500 && k1 < 500) ? mask[((size_t)b * 500 + rg) * 500 + k1] : 0.f;
                float m2 = (rg < 500 && k2 < 500) ? mask[((size_t)b * 500 + rg) * 500 + k2] : 0.f;
                v1[r] = (k1 < 500) ? s1[r] + m1 : -1e30f;
                v2[r] = (k2 < 500) ? s2[r] + m2 : -1e30f;
            }
#pragma unroll
            for (int r = 0; r < 8; ++r) {
                float cm = fmaxf(v1[r], v2[r]);
#pragma unroll
                for (int d = 1; d < 16; d <<= 1) cm = fmaxf(cm, __shfl_xor(cm, d, 16));
                float nm = fmaxf(mrun[r], cm);
                float sc = __expf(mrun[r] - nm);
                acc[r] *= sc;
                float p1 = __expf(v1[r] - nm), p2 = __expf(v2[r] - nm);
                float ls = p1 + p2;
#pragma unroll
                for (int d = 1; d < 16; d <<= 1) ls += __shfl_xor(ls, d, 16);
                lrun[r] = lrun[r] * sc + ls;
                mrun[r] = nm;
                sP[w][r + (hi << 3)][cr]      = (half_t)p1;
                sP[w][r + (hi << 3)][16 + cr] = (half_t)p2;
            }
            v16h aP = make_a(*(const v8h*)&sP[w][cr][kb], *(const v8h*)&sP[w][cr][kb + 16]);
            v16h bV = make_a(*(const v8h*)&sVT[cr][kc + kb], *(const v8h*)&sVT[cr][kc + kb + 16]);
            acc = WMMA_F16(aP, bV, acc);
        }
#pragma unroll
        for (int r = 0; r < 8; ++r) {
            const int nq = n0 + r + (hi << 3);
            if (nq < 500)
                outc[(size_t)(b * 500 + nq) * 128 + h * 16 + cr] = (half_t)(acc[r] / lrun[r]);
        }
    }
}

// ---------------------------------------------------------------------------
// Kernel 3: mh = outc@Wc + bc, s = mh@nodes^T/sqrt(128), 10*tanh + mask,
// row softmax -> probs.  One block per (b, 16-query tile).
// ---------------------------------------------------------------------------
__global__ __launch_bounds__(256) void combine_kernel(
    const half_t* __restrict__ outc, const half_t* __restrict__ pWc,
    const float* __restrict__ bc, const half_t* __restrict__ Nh,
    const float* __restrict__ mask, float* __restrict__ probs) {
    __shared__ __align__(16) half_t sMH[16][136];
    __shared__ __align__(16) float  sS[16][520];
    const int b = blockIdx.x >> 5, nt = blockIdx.x & 31;
    const int n0 = nt * 16;
    const int lane = threadIdx.x & 31, w = threadIdx.x >> 5;
    const int cr = lane & 15, hi = lane >> 4, kb = hi << 3;
    v8h z8 = {};
    // --- stage 1: mh column tile [16w, 16w+16) ---
    v8f acc = {};
    const int arow = n0 + cr;
    const bool av = arow < 500;
#pragma unroll
    for (int kc = 0; kc < 4; ++kc) {
        v16h a;
        if (av) {
            const half_t* ap = outc + (size_t)(b * 500 + arow) * 128 + kc * 32 + kb;
            a = make_a(*(const v8h*)ap, *(const v8h*)(ap + 16));
        } else a = make_a(z8, z8);
        v16h bb = *(const v16h*)(pWc + (w * 4 + kc) * 512 + lane * 16);
        acc = WMMA_F16(a, bb, acc);
    }
    const float bias = bc[w * 16 + cr];
#pragma unroll
    for (int r = 0; r < 8; ++r) sMH[r + (hi << 3)][w * 16 + cr] = (half_t)(acc[r] + bias);
    __syncthreads();
    // --- stage 2: pointer logits over all keys ---
    const float inv_se = 1.0f / 11.313708498984761f;
    for (int mt = w; mt < 32; mt += 8) {
        const int m0 = mt * 16;
        v8f sa = {};
#pragma unroll
        for (int kc = 0; kc < 4; ++kc) {
            v16h a = make_a(*(const v8h*)&sMH[cr][kc * 32 + kb],
                            *(const v8h*)&sMH[cr][kc * 32 + kb + 16]);
            const int mr = m0 + cr;
            v16h bb;
            if (mr < 500) {
                const half_t* np = Nh + (size_t)(b * 500 + mr) * 128 + kc * 32 + kb;
                bb = make_a(*(const v8h*)np, *(const v8h*)(np + 16));
            } else bb = make_a(z8, z8);
            sa = WMMA_F16(a, bb, sa);
        }
#pragma unroll
        for (int r = 0; r < 8; ++r) {
            const int row = r + (hi << 3);
            const int nq = n0 + row, mk = m0 + cr;
            float s = 10.0f * tanhf(sa[r] * inv_se);
            if (nq < 500 && mk < 500) s += mask[((size_t)b * 500 + nq) * 500 + mk];
            if (mk >= 500) s = -1e30f;
            sS[row][mk] = s;
        }
    }
    __syncthreads();
    // --- stage 3: softmax per row, 16 threads/row ---
    const int row = threadIdx.x >> 4, sub = threadIdx.x & 15;
    float mx = -1e30f;
#pragma unroll 4
    for (int j = 0; j < 32; ++j) mx = fmaxf(mx, sS[row][j * 16 + sub]);
#pragma unroll
    for (int d = 1; d < 16; d <<= 1) mx = fmaxf(mx, __shfl_xor(mx, d, 16));
    float sum = 0.f;
#pragma unroll 4
    for (int j = 0; j < 32; ++j) {
        float e = __expf(sS[row][j * 16 + sub] - mx);
        sS[row][j * 16 + sub] = e;
        sum += e;
    }
#pragma unroll
    for (int d = 1; d < 16; d <<= 1) sum += __shfl_xor(sum, d, 16);
    const float inv = 1.0f / sum;
    const int nq = n0 + row;
    if (nq < 500) {
        float* op = probs + ((size_t)b * 500 + nq) * 500;
#pragma unroll 4
        for (int j = 0; j < 32; ++j) {
            const int mm = j * 16 + sub;
            if (mm < 500) op[mm] = sS[row][mm] * inv;
        }
    }
}

extern "C" void kernel_launch(void* const* d_in, const int* in_sizes, int n_in,
                              void* d_out, int out_size, void* d_ws, size_t ws_size,
                              hipStream_t stream) {
    const float* nodes = (const float*)d_in[0];
    const float* q1    = (const float*)d_in[1];
    const float* q0    = (const float*)d_in[2];
    const float* mask  = (const float*)d_in[3];
    const float* Wq1   = (const float*)d_in[4];
    const float* Wq0   = (const float*)d_in[5];
    const float* Wk    = (const float*)d_in[6];
    const float* Wv    = (const float*)d_in[7];
    const float* Wc    = (const float*)d_in[8];
    const float* bcp   = (const float*)d_in[9];
    float* probs = (float*)d_out;

    // Workspace layout (f16): 5 tensors of 32000x128 + 5 packed weights -> ~41 MB
    const size_t RE = (size_t)32000 * 128;
    half_t* ws = (half_t*)d_ws;
    half_t* Kh = ws;
    half_t* Vh = Kh + RE;
    half_t* Qh = Vh + RE;
    half_t* Nh = Qh + RE;
    half_t* OC = Nh + RE;
    half_t* PW = OC + RE;                 // 5 * 16384 halves
    half_t* pWq1 = PW;
    half_t* pWq0 = PW + 16384;
    half_t* pWk  = PW + 32768;
    half_t* pWv  = PW + 49152;
    half_t* pWc  = PW + 65536;

    pack_weights<<<320, 256, 0, stream>>>(Wq1, Wq0, Wk, Wv, Wc, PW);
    proj_kernel<<<2000, 256, 0, stream>>>(nodes, q1, q0, pWk, pWv, pWq1, pWq0, Kh, Vh, Qh, Nh);
    attn_kernel<<<512, 256, 0, stream>>>(Kh, Vh, Qh, mask, OC);
    combine_kernel<<<2048, 256, 0, stream>>>(OC, pWc, bcp, Nh, mask, probs);
}
